// GraphAttentionBlock_65481071395687
// MI455X (gfx1250) — compile-verified, matrix operand order checked by reading, not run on hardware
//
#include <hip/hip_runtime.h>

// GraphAttentionBlock for MI455X (gfx1250, wave32).
// bf16 WMMA (v_wmma_f32_16x16x32_bf16, f32 accum) for all matmuls.
// Flash-attention streaming softmax (never materializes the 512MB score tensor;
// adjacency's 64MB int32 read is the unavoidable HBM floor ~3us @ 23.3TB/s).
// K/V chunks are staged to LDS with CDNA5 async-to-LDS loads (ASYNCcnt),
// double-buffered so the DMA of chunk i+1 overlaps compute of chunk i.

#define NN 4096
#define EE 256
#define HH 8
#define DD 32

typedef __attribute__((ext_vector_type(16))) __bf16 v16bf;
typedef __attribute__((ext_vector_type(8)))  __bf16 bf16x8;
typedef __attribute__((ext_vector_type(8)))  float  v8f;
typedef __attribute__((ext_vector_type(4)))  float  f32x4;

// ---- CDNA5 async-to-LDS (ISA 15.18.3 op 96..98, ASYNCcnt) ------------------
__device__ __forceinline__ void async_b128_to_lds(unsigned lds_addr, const void* gaddr) {
  asm volatile("global_load_async_to_lds_b128 %0, %1, off"
               :: "v"(lds_addr), "v"((unsigned long long)(size_t)gaddr)
               : "memory");
}
__device__ __forceinline__ void wait_async0() {
  asm volatile("s_wait_asynccnt 0x0" ::: "memory");
}
// Low 32 bits of a generic pointer into LDS == DS byte address (ISA 10.2).
__device__ __forceinline__ unsigned lds_addr_of(const void* p) {
  return (unsigned)(unsigned long long)(size_t)p;
}

// ---- fragment helpers (ISA 7.12.2 16-bit layouts) --------------------------
// lane l -> row/col (l & 15), half = l>>4
// element e -> K = e + 8*half (e<8), K = 16 + 8*half + (e-8) (e>=8)

__device__ __forceinline__ v16bf load_frag_f32(const float* __restrict__ base,
                                               int ld, int rc0, int k0, int lane) {
  int rc = rc0 + (lane & 15);
  int half = lane >> 4;
  const float* p = base + (size_t)rc * ld + k0 + half * 8;
  f32x4 a0 = *(const f32x4*)(p);
  f32x4 a1 = *(const f32x4*)(p + 4);
  f32x4 b0 = *(const f32x4*)(p + 16);
  f32x4 b1 = *(const f32x4*)(p + 20);
  v16bf v;
#pragma unroll
  for (int i = 0; i < 4; ++i) {
    v[i]      = (__bf16)a0[i];
    v[4 + i]  = (__bf16)a1[i];
    v[8 + i]  = (__bf16)b0[i];
    v[12 + i] = (__bf16)b1[i];
  }
  return v;
}

__device__ __forceinline__ v16bf load_frag_bf16(const __bf16* __restrict__ base,
                                                int ld, int rc0, int k0, int lane) {
  int rc = rc0 + (lane & 15);
  int half = lane >> 4;
  const __bf16* p = base + (size_t)rc * ld + k0 + half * 8;
  bf16x8 g0 = *(const bf16x8*)(p);
  bf16x8 g1 = *(const bf16x8*)(p + 16);
  v16bf v;
#pragma unroll
  for (int i = 0; i < 8; ++i) { v[i] = g0[i]; v[8 + i] = g1[i]; }
  return v;
}

// Fragment whose per-lane data is a contiguous row in a [32][32] LDS tile
// (A layout, or B = tile^T): 2 x ds_load_b128 per lane.
__device__ __forceinline__ v16bf frag_lds_rows(const __bf16* tile, int r0, int lane) {
  int r = r0 + (lane & 15);
  int half = lane >> 4;
  const __bf16* p = tile + r * 32 + half * 8;
  bf16x8 g0 = *(const bf16x8*)(p);
  bf16x8 g1 = *(const bf16x8*)(p + 16);
  v16bf v;
#pragma unroll
  for (int i = 0; i < 8; ++i) { v[i] = g0[i]; v[8 + i] = g1[i]; }
  return v;
}

// B[k][n] fragment gathered down a column of a [32][32] LDS tile (used for V).
__device__ __forceinline__ v16bf fragB_lds_cols(const __bf16* tile, int col0, int lane) {
  int c = col0 + (lane & 15);
  int half = lane >> 4;
  v16bf v;
#pragma unroll
  for (int e = 0; e < 8; ++e)
    v[e] = tile[(half * 8 + e) * 32 + c];
#pragma unroll
  for (int e = 0; e < 8; ++e)
    v[8 + e] = tile[(16 + half * 8 + e) * 32 + c];
  return v;
}

// ---- kernel 1: fused QKV projection ----------------------------------------
__global__ __launch_bounds__(128) void qkv_proj_kernel(
    const float* __restrict__ x,
    const float* __restrict__ Wq, const float* __restrict__ bq,
    const float* __restrict__ Wk, const float* __restrict__ bk,
    const float* __restrict__ Wv, const float* __restrict__ bv,
    __bf16* __restrict__ Qb, __bf16* __restrict__ Kb, __bf16* __restrict__ Vb) {
  int lane = threadIdx.x & 31;
  int wave = threadIdx.x >> 5;
  int rowt = blockIdx.x * 16;
  int grp  = blockIdx.y;
  int mat  = grp >> 2;
  int col0 = (grp & 3) * 64 + wave * 16;

  const float* W;
  const float* bias;
  __bf16* out;
  float oscale;
  if (mat == 0)      { W = Wq; bias = bq; out = Qb; oscale = 0.17677669529663687f; } // 1/sqrt(32)
  else if (mat == 1) { W = Wk; bias = bk; out = Kb; oscale = 1.0f; }
  else               { W = Wv; bias = bv; out = Vb; oscale = 1.0f; }

  v8f acc = {};
#pragma unroll
  for (int k0 = 0; k0 < EE; k0 += 32) {
    v16bf a = load_frag_f32(x, EE, rowt, k0, lane);   // A: x rows
    v16bf b = load_frag_f32(W, EE, col0, k0, lane);   // B: W^T
    acc = __builtin_amdgcn_wmma_f32_16x16x32_bf16(false, a, false, b,
                                                  (short)0, acc, false, false);
  }
  int col = col0 + (lane & 15);
  int half = lane >> 4;
  float bb = bias[col];
#pragma unroll
  for (int r = 0; r < 8; ++r) {
    int row = rowt + r + 8 * half;
    out[(size_t)row * EE + col] = (__bf16)((acc[r] + bb) * oscale);
  }
}

// ---- kernel 2: flash attention with adjacency mask -------------------------
// grid (N/64, H), 4 waves/block; each wave owns a 16-row tile of one head.
// K/V 32x32 bf16 chunks staged once per block via async-to-LDS, double-buffered.
__global__ __launch_bounds__(128) void flash_attn_kernel(
    const __bf16* __restrict__ Qb, const __bf16* __restrict__ Kb,
    const __bf16* __restrict__ Vb, const int* __restrict__ adj,
    __bf16* __restrict__ Ab) {
  __shared__ float  s_scores[4][16][33];
  __shared__ __bf16 s_p[4][16][32];
  __shared__ float  s_fac[4][16];
  __shared__ float  s_stat[4][16];
  __shared__ __bf16 s_k[2][32][32];   // double-buffered K chunk (rows m, cols d)
  __shared__ __bf16 s_v[2][32][32];   // double-buffered V chunk

  int lane = threadIdx.x & 31;
  int wave = threadIdx.x >> 5;
  int lrow = lane & 15;
  int half = lane >> 4;
  int h  = blockIdx.y;
  int nt = blockIdx.x * 64 + wave * 16;
  int hd = h * DD;

  // staging role: thread t loads one b128 per matrix per chunk
  int t    = threadIdx.x;
  int srow = t >> 2;           // 0..31
  int scol = (t & 3) * 8;      // 0,8,16,24

  // Q fragment fixed for the whole sweep (1/sqrt(D) folded into Qb).
  v16bf qa = load_frag_bf16(Qb, EE, nt, hd, lane);

  // prologue: stage chunk 0 into buffer 0
  async_b128_to_lds(lds_addr_of(&s_k[0][srow][scol]),
                    Kb + (size_t)srow * EE + hd + scol);
  async_b128_to_lds(lds_addr_of(&s_v[0][srow][scol]),
                    Vb + (size_t)srow * EE + hd + scol);
  wait_async0();
  __syncthreads();

  v8f acc0 = {}, acc1 = {};
  float m_i = -3.0e38f, l_i = 0.0f;

  for (int mc = 0; mc < NN; mc += 32) {
    int bi = (mc >> 5) & 1;

    // prefetch chunk mc+32 into the other buffer (overlaps all compute below)
    if (mc + 32 < NN) {
      async_b128_to_lds(lds_addr_of(&s_k[bi ^ 1][srow][scol]),
                        Kb + (size_t)(mc + 32 + srow) * EE + hd + scol);
      async_b128_to_lds(lds_addr_of(&s_v[bi ^ 1][srow][scol]),
                        Vb + (size_t)(mc + 32 + srow) * EE + hd + scol);
    }

    // S tile: two 16x16 WMMAs over K = D = 32, B = K^T from LDS rows
    v16bf kb0 = frag_lds_rows(&s_k[bi][0][0], 0, lane);
    v16bf kb1 = frag_lds_rows(&s_k[bi][0][0], 16, lane);
    v8f z = {};
    v8f s0 = __builtin_amdgcn_wmma_f32_16x16x32_bf16(false, qa, false, kb0,
                                                     (short)0, z, false, false);
    v8f s1 = __builtin_amdgcn_wmma_f32_16x16x32_bf16(false, qa, false, kb1,
                                                     (short)0, z, false, false);
#pragma unroll
    for (int r = 0; r < 8; ++r) {
      s_scores[wave][r + 8 * half][lrow]      = s0[r];
      s_scores[wave][r + 8 * half][lrow + 16] = s1[r];
    }
    __syncthreads();

    // masked online softmax; lane i (i<16) owns row i of the tile.
    if (lane < 16) {
      const int* arow = adj + (size_t)(nt + lane) * NN + mc;
      float sv[32];
      float cmax = -3.0e38f;
#pragma unroll
      for (int j = 0; j < 32; ++j) {
        float s = s_scores[wave][lane][j];
        s = (arow[j] != 0) ? s : -3.0e38f;
        sv[j] = s;
        cmax = fmaxf(cmax, s);
      }
      float m_new = fmaxf(m_i, cmax);
      float f = __expf(m_i - m_new);
      float psum = 0.0f;
#pragma unroll
      for (int j = 0; j < 32; ++j) {
        float p = (sv[j] <= -1.0e38f) ? 0.0f : __expf(sv[j] - m_new);
        psum += p;
        s_p[wave][lane][j] = (__bf16)p;
      }
      l_i = l_i * f + psum;
      m_i = m_new;
      s_fac[wave][lane] = f;
    }
    __syncthreads();

    // rescale running output accumulators by per-row factor
#pragma unroll
    for (int r = 0; r < 8; ++r) {
      float f = s_fac[wave][r + 8 * half];
      acc0[r] *= f;
      acc1[r] *= f;
    }

    // P fragment (A layout) from LDS -> layout transpose
    v16bf pa;
    {
      const __bf16* pp = &s_p[wave][lrow][0];
#pragma unroll
      for (int e = 0; e < 8; ++e) {
        pa[e]     = pp[half * 8 + e];
        pa[8 + e] = pp[16 + half * 8 + e];
      }
    }

    // O += P @ V : two 16x16 WMMAs, V B-fragments gathered from LDS columns
    v16bf vb0 = fragB_lds_cols(&s_v[bi][0][0], 0, lane);
    v16bf vb1 = fragB_lds_cols(&s_v[bi][0][0], 16, lane);
    acc0 = __builtin_amdgcn_wmma_f32_16x16x32_bf16(false, pa, false, vb0,
                                                   (short)0, acc0, false, false);
    acc1 = __builtin_amdgcn_wmma_f32_16x16x32_bf16(false, pa, false, vb1,
                                                   (short)0, acc1, false, false);

    // drain this wave's async stores, then publish next chunk block-wide;
    // also protects s_p/s_scores/s_k/s_v before the next iteration.
    wait_async0();
    __syncthreads();
  }

  if (lane < 16) s_stat[wave][lane] = (l_i > 0.0f) ? (1.0f / l_i) : 0.0f;
  __syncthreads();
#pragma unroll
  for (int r = 0; r < 8; ++r) {
    int row = nt + r + 8 * half;
    float linv = s_stat[wave][r + 8 * half];
    Ab[(size_t)row * EE + hd + lrow]      = (__bf16)(acc0[r] * linv);
    Ab[(size_t)row * EE + hd + 16 + lrow] = (__bf16)(acc1[r] * linv);
  }
}

// ---- kernel 3: output projection -------------------------------------------
__global__ __launch_bounds__(128) void out_proj_kernel(
    const __bf16* __restrict__ Ab, const float* __restrict__ Wo,
    const float* __restrict__ bo, float* __restrict__ out) {
  int lane = threadIdx.x & 31;
  int wave = threadIdx.x >> 5;
  int rowt = blockIdx.x * 16;
  int col0 = blockIdx.y * 64 + wave * 16;

  v8f acc = {};
#pragma unroll
  for (int k0 = 0; k0 < EE; k0 += 32) {
    v16bf a = load_frag_bf16(Ab, EE, rowt, k0, lane);
    v16bf b = load_frag_f32(Wo, EE, col0, k0, lane);
    acc = __builtin_amdgcn_wmma_f32_16x16x32_bf16(false, a, false, b,
                                                  (short)0, acc, false, false);
  }
  int col = col0 + (lane & 15);
  int half = lane >> 4;
  float bb = bo[col];
#pragma unroll
  for (int r = 0; r < 8; ++r)
    out[(size_t)(rowt + r + 8 * half) * EE + col] = acc[r] + bb;
}

// ---- launch -----------------------------------------------------------------
extern "C" void kernel_launch(void* const* d_in, const int* in_sizes, int n_in,
                              void* d_out, int out_size, void* d_ws, size_t ws_size,
                              hipStream_t stream) {
  const float* x   = (const float*)d_in[0];
  const int*   adj = (const int*)d_in[1];
  const float* Wq  = (const float*)d_in[2];
  const float* bq  = (const float*)d_in[3];
  const float* Wk  = (const float*)d_in[4];
  const float* bk  = (const float*)d_in[5];
  const float* Wv  = (const float*)d_in[6];
  const float* bv  = (const float*)d_in[7];
  const float* Wo  = (const float*)d_in[8];
  const float* bo  = (const float*)d_in[9];
  float* out = (float*)d_out;

  // workspace: 4 bf16 [N][E] buffers = 8 MB total
  __bf16* Qb = (__bf16*)d_ws;
  __bf16* Kb = Qb + (size_t)NN * EE;
  __bf16* Vb = Kb + (size_t)NN * EE;
  __bf16* Ab = Vb + (size_t)NN * EE;

  qkv_proj_kernel<<<dim3(NN / 16, 12), 128, 0, stream>>>(
      x, Wq, bq, Wk, bk, Wv, bv, Qb, Kb, Vb);
  flash_attn_kernel<<<dim3(NN / 64, HH), 128, 0, stream>>>(Qb, Kb, Vb, adj, Ab);
  out_proj_kernel<<<dim3(NN / 16, EE / 64), 128, 0, stream>>>(Ab, Wo, bo, out);
}
